// CodeBook_29738353557976
// MI455X (gfx1250) — matrix operand order repaired and checked by reading, not compiled, not run on hardware
//
#include <hip/hip_runtime.h>

#define DIMD     128
#define KCODES   512
#define ROW_PAD  136              // 128 + 8 halves of padding -> conflict-free LDS fragment loads
#define MTILE    256              // x-rows per block (8 waves x 32 rows)
#define NTHREADS 256

typedef _Float16 half_t;
typedef __attribute__((ext_vector_type(16))) half_t v16h;
typedef __attribute__((ext_vector_type(8)))  half_t v8h;
typedef __attribute__((ext_vector_type(8)))  float  v8f;

// ---------------------------------------------------------------------------
// Kernel A: normalize codebook rows, split into f16 hi/lo planes (padded
// stride), and record ||cn||^2 per code.  hi+lo carries ~22 mantissa bits,
// so 3 f16 WMMAs reproduce the f32 GEMM to ~2^-22 relative error.
// ---------------------------------------------------------------------------
__global__ void vq_cbprep(const float* __restrict__ cb,
                          half_t* __restrict__ wsHi, half_t* __restrict__ wsLo,
                          float* __restrict__ wsNorm)
{
    int r = blockIdx.x * blockDim.x + threadIdx.x;
    if (r >= KCODES) return;
    const float* row = cb + r * DIMD;
    float ss = 0.f;
    #pragma unroll 8
    for (int k = 0; k < DIMD; ++k) { float v = row[k]; ss = fmaf(v, v, ss); }
    float scale = 1.0f / fmaxf(sqrtf(ss), 1e-12f);   // matches F.normalize eps semantics
    float n2 = 0.f;
    half_t* hi = wsHi + r * ROW_PAD;
    half_t* lo = wsLo + r * ROW_PAD;
    #pragma unroll 4
    for (int k = 0; k < DIMD; ++k) {
        float cn = row[k] * scale;
        half_t h = (half_t)cn;
        hi[k] = h;
        lo[k] = (half_t)(cn - (float)h);
        n2 = fmaf(cn, cn, n2);
    }
    wsNorm[r] = n2;
}

// ---------------------------------------------------------------------------
// Kernel B: main VQ search + gather + STE output + per-block loss partials.
// Whole split codebook resident in LDS; each wave owns 32 rows so every
// B fragment load feeds 24 WMMAs (LDS traffic per MAC halved vs 16 rows).
// ---------------------------------------------------------------------------
__global__ __launch_bounds__(NTHREADS) void vq_main(
    const float* __restrict__ x, const float* __restrict__ cb,
    const half_t* __restrict__ wsHi, const half_t* __restrict__ wsLo,
    const float* __restrict__ wsNorm,
    float* __restrict__ outQ, float* __restrict__ outIdx,
    float* __restrict__ partials, int nRows)
{
    __shared__ half_t sHi[KCODES * ROW_PAD];
    __shared__ half_t sLo[KCODES * ROW_PAD];
    __shared__ float  sNorm[KCODES];
    __shared__ int    sIdx[MTILE];
    __shared__ float  sRed[NTHREADS];

    const int tid = threadIdx.x;

    // ---- stage split codebook into LDS (16B vector copies) ----
    {
        const uint4* gh = (const uint4*)wsHi;
        const uint4* gl = (const uint4*)wsLo;
        uint4* lh = (uint4*)sHi;
        uint4* ll = (uint4*)sLo;
        const int n16 = KCODES * ROW_PAD / 8;     // 8704 x 16B
        for (int i = tid; i < n16; i += NTHREADS) { lh[i] = gh[i]; ll[i] = gl[i]; }
        for (int i = tid; i < KCODES; i += NTHREADS) sNorm[i] = wsNorm[i];
    }
    __syncthreads();

    const int lane = tid & 31;
    const int wave = tid >> 5;
    const int h    = lane >> 4;          // K-half selector for fragments
    const int mloc = lane & 15;          // row within M-tile / code column in tile
    const int rowBase = blockIdx.x * MTILE + wave * 32;

    // ---- build f16 hi/lo A fragments for this wave's two M-tiles ----
    v16h ahi0[4], alo0[4], ahi1[4], alo1[4];
    #pragma unroll
    for (int half = 0; half < 2; ++half) {
        const int grow = rowBase + half * 16 + mloc;
        float xa[64];
        #pragma unroll
        for (int i = 0; i < 64; ++i) xa[i] = 0.f;
        if (grow < nRows) {
            const float4* rp = (const float4*)(x + (size_t)grow * DIMD);
            #pragma unroll
            for (int j = 0; j < 8; ++j) {          // cols 8h + 16j + i
                float4 a = rp[4 * j + 2 * h];
                float4 b = rp[4 * j + 2 * h + 1];
                xa[8*j+0]=a.x; xa[8*j+1]=a.y; xa[8*j+2]=a.z; xa[8*j+3]=a.w;
                xa[8*j+4]=b.x; xa[8*j+5]=b.y; xa[8*j+6]=b.z; xa[8*j+7]=b.w;
            }
        }
        float ss = 0.f;
        #pragma unroll
        for (int i = 0; i < 64; ++i) ss = fmaf(xa[i], xa[i], ss);
        ss += __shfl_xor(ss, 16, 32);    // lanes L, L^16 hold complementary halves of a row
        const float scale = 1.0f / fmaxf(sqrtf(ss), 1e-12f);
        v16h* dsthi = half ? ahi1 : ahi0;
        v16h* dstlo = half ? alo1 : alo0;
        #pragma unroll
        for (int s = 0; s < 4; ++s) {
            #pragma unroll
            for (int e = 0; e < 16; ++e) {
                float cn = xa[16 * s + e] * scale;
                half_t hv = (half_t)cn;
                dsthi[s][e] = hv;
                dstlo[s][e] = (half_t)(cn - (float)hv);
            }
        }
    }

    // ---- sweep all 512 codes: 32 tiles x (4 ksteps x 6 WMMA) ----
    float bs0[8], bs1[8]; int bi0[8], bi1[8];
    #pragma unroll
    for (int r = 0; r < 8; ++r) {
        bs0[r] = 3.4e38f; bi0[r] = 0;
        bs1[r] = 3.4e38f; bi1[r] = 0;
    }

    for (int t = 0; t < KCODES / 16; ++t) {
        const int code = t * 16 + mloc;
        const half_t* bhp = sHi + code * ROW_PAD + 8 * h;
        const half_t* blp = sLo + code * ROW_PAD + 8 * h;
        v8f c0 = {};
        v8f c1 = {};
        #pragma unroll
        for (int s = 0; s < 4; ++s) {
            v8h h0 = *(const v8h*)(bhp + 32 * s);
            v8h h1 = *(const v8h*)(bhp + 32 * s + 16);
            v8h l0 = *(const v8h*)(blp + 32 * s);
            v8h l1 = *(const v8h*)(blp + 32 * s + 16);
            v16h bhi = __builtin_shufflevector(h0, h1, 0,1,2,3,4,5,6,7,8,9,10,11,12,13,14,15);
            v16h blo = __builtin_shufflevector(l0, l1, 0,1,2,3,4,5,6,7,8,9,10,11,12,13,14,15);
            c0 = __builtin_amdgcn_wmma_f32_16x16x32_f16(false, ahi0[s], false, bhi, (short)0, c0, false, false);
            c1 = __builtin_amdgcn_wmma_f32_16x16x32_f16(false, ahi1[s], false, bhi, (short)0, c1, false, false);
            c0 = __builtin_amdgcn_wmma_f32_16x16x32_f16(false, ahi0[s], false, blo, (short)0, c0, false, false);
            c1 = __builtin_amdgcn_wmma_f32_16x16x32_f16(false, ahi1[s], false, blo, (short)0, c1, false, false);
            c0 = __builtin_amdgcn_wmma_f32_16x16x32_f16(false, alo0[s], false, bhi, (short)0, c0, false, false);
            c1 = __builtin_amdgcn_wmma_f32_16x16x32_f16(false, alo1[s], false, bhi, (short)0, c1, false, false);
        }
        const float cn2 = sNorm[code];   // ||xn||^2 is constant per row -> irrelevant for argmin
        #pragma unroll
        for (int r = 0; r < 8; ++r) {
            float sc0 = fmaf(-2.0f, c0[r], cn2);
            float sc1 = fmaf(-2.0f, c1[r], cn2);
            if (sc0 < bs0[r]) { bs0[r] = sc0; bi0[r] = code; }
            if (sc1 < bs1[r]) { bs1[r] = sc1; bi1[r] = code; }
        }
    }

    // ---- reduce argmin across the 16 lanes of each half (smaller index wins ties) ----
    #pragma unroll
    for (int m = 1; m <= 8; m <<= 1) {
        #pragma unroll
        for (int r = 0; r < 8; ++r) {
            float os0 = __shfl_xor(bs0[r], m, 32);
            int   oi0 = __shfl_xor(bi0[r], m, 32);
            if (os0 < bs0[r] || (os0 == bs0[r] && oi0 < bi0[r])) { bs0[r] = os0; bi0[r] = oi0; }
            float os1 = __shfl_xor(bs1[r], m, 32);
            int   oi1 = __shfl_xor(bi1[r], m, 32);
            if (os1 < bs1[r] || (os1 == bs1[r] && oi1 < bi1[r])) { bs1[r] = os1; bi1[r] = oi1; }
        }
    }
    if (mloc == 0) {
        #pragma unroll
        for (int r = 0; r < 8; ++r) {
            int m0 = 8 * h + r;          // C/D layout: VGPR r <-> M = r + 8*(lane>=16)
            int g0 = rowBase + m0;
            if (g0 < nRows) {
                sIdx[wave * 32 + m0] = bi0[r];
                outIdx[g0] = (float)bi0[r];
            }
            int g1 = rowBase + 16 + m0;
            if (g1 < nRows) {
                sIdx[wave * 32 + 16 + m0] = bi1[r];
                outIdx[g1] = (float)bi1[r];
            }
        }
    }
    __syncthreads();

    // ---- gather original codebook rows, STE output, loss partials ----
    float acc = 0.f;
    const int blk0 = blockIdx.x * MTILE;
    for (int i = tid; i < MTILE * (DIMD / 4); i += NTHREADS) {
        int lr = i >> 5;                 // DIMD/4 == 32 float4 per row
        int cc = i & 31;
        int gr = blk0 + lr;
        if (gr < nRows) {
            float4 q  = ((const float4*)(cb + sIdx[lr] * DIMD))[cc];
            float4 xv = ((const float4*)(x + (size_t)gr * DIMD))[cc];
            float dx = q.x - xv.x, dy = q.y - xv.y, dz = q.z - xv.z, dw = q.w - xv.w;
            float4 o; o.x = xv.x + dx; o.y = xv.y + dy; o.z = xv.z + dz; o.w = xv.w + dw;
            ((float4*)(outQ + (size_t)gr * DIMD))[cc] = o;
            acc = fmaf(dx, dx, acc); acc = fmaf(dy, dy, acc);
            acc = fmaf(dz, dz, acc); acc = fmaf(dw, dw, acc);
        }
    }
    sRed[tid] = acc;
    __syncthreads();
    for (int ofs = NTHREADS / 2; ofs > 0; ofs >>= 1) {
        if (tid < ofs) sRed[tid] += sRed[tid + ofs];
        __syncthreads();
    }
    if (tid == 0) partials[blockIdx.x] = sRed[0];
}

// ---------------------------------------------------------------------------
// Kernel C: deterministic final loss reduction (both losses are numerically
// the same value: mean((q-x)^2) == mean((x-q)^2)).
// ---------------------------------------------------------------------------
__global__ void vq_reduce(const float* __restrict__ partials, int nb,
                          float* __restrict__ scalars, double invCount)
{
    __shared__ double s[NTHREADS];
    double a = 0.0;
    for (int i = threadIdx.x; i < nb; i += NTHREADS) a += (double)partials[i];
    s[threadIdx.x] = a;
    __syncthreads();
    for (int ofs = NTHREADS / 2; ofs > 0; ofs >>= 1) {
        if (threadIdx.x < ofs) s[threadIdx.x] += s[threadIdx.x + ofs];
        __syncthreads();
    }
    if (threadIdx.x == 0) {
        float m = (float)(s[0] * invCount);
        scalars[0] = m;   // codebook_loss
        scalars[1] = m;   // commitment_loss
    }
}

// ---------------------------------------------------------------------------
extern "C" void kernel_launch(void* const* d_in, const int* in_sizes, int n_in,
                              void* d_out, int out_size, void* d_ws, size_t ws_size,
                              hipStream_t stream)
{
    const float* x  = (const float*)d_in[0];
    const float* cb = (const float*)d_in[1];
    const int nRows = in_sizes[0] / DIMD;     // 500000

    float* out     = (float*)d_out;
    float* outQ    = out;                              // [N*D] quantized_ste
    float* scalars = out + (size_t)nRows * DIMD;       // [2] losses
    float* outIdx  = scalars + 2;                      // [N] indices (as float)

    char*   ws     = (char*)d_ws;
    half_t* wsHi   = (half_t*)ws;                                    // 139264 B
    half_t* wsLo   = (half_t*)(ws + (size_t)KCODES * ROW_PAD * 2);   // 139264 B
    float*  wsNorm = (float*)(ws + (size_t)KCODES * ROW_PAD * 4);    // 2048 B
    float*  parts  = (float*)(ws + (size_t)KCODES * ROW_PAD * 4 + KCODES * 4);

    const int nBlocks = (nRows + MTILE - 1) / MTILE;

    vq_cbprep<<<(KCODES + NTHREADS - 1) / NTHREADS, NTHREADS, 0, stream>>>(
        cb, wsHi, wsLo, wsNorm);
    vq_main<<<nBlocks, NTHREADS, 0, stream>>>(
        x, cb, wsHi, wsLo, wsNorm, outQ, outIdx, parts, nRows);
    vq_reduce<<<1, NTHREADS, 0, stream>>>(
        parts, nBlocks, scalars, 1.0 / ((double)nRows * DIMD));
}
